// CrossAttention_6880537608752
// MI455X (gfx1250) — compile-verified, matrix operand order checked by reading, not run on hardware
//
#include <hip/hip_runtime.h>

typedef __attribute__((ext_vector_type(16))) _Float16 v16h;
typedef __attribute__((ext_vector_type(8)))  _Float16 v8h;
typedef __attribute__((ext_vector_type(8)))  float    v8f;

#define kTB 8
#define kNQ 256
#define kNK 4096
#define kC  256
#define kH  8
#define kDH 32

// ---------------------------------------------------------------- helpers ---

__device__ __forceinline__ v16h frag_combine(v8h lo, v8h hi) {
  v16h r;
#pragma unroll
  for (int i = 0; i < 8; ++i) { r[i] = lo[i]; r[8 + i] = hi[i]; }
  return r;
}

// Load a 16x32 f16 WMMA fragment (A/B symmetric per-lane layout) from
// row-major storage: element (row, k) at base[row*stride + k].
// Per ISA: lane (row = l&15, hi = l>>4): halves 0..7 <- k = hi*8+0..7,
// halves 8..15 <- k = 16 + hi*8 + 0..7 -> two contiguous 16B loads.
__device__ __forceinline__ v16h load_frag(const _Float16* base, int stride) {
  int lane = threadIdx.x & 31;
  int r = lane & 15, hi = lane >> 4;
  const _Float16* p = base + (size_t)r * stride + hi * 8;
  v8h lo = *(const v8h*)p;
  v8h h2 = *(const v8h*)(p + 16);
  return frag_combine(lo, h2);
}

__device__ __forceinline__ float spike4(float x) {  // quant4(x) * 0.25
  return rintf(fminf(fmaxf(x, 0.f), 4.f)) * 0.25f;
}

// ---------------------------------------------------- weight fold (s*W, b) ---

__global__ void prep_weights(const float* __restrict__ W, const float* __restrict__ b,
                             const float* __restrict__ s, const float* __restrict__ off,
                             _Float16* __restrict__ We, float* __restrict__ be) {
  int idx = blockIdx.x * 256 + threadIdx.x;   // 65536 elements
  int o = idx >> 8, c = idx & 255;
  We[idx] = (_Float16)(s[o] * W[idx]);
  if (c == 0) be[o] = s[o] * b[o] + off[o];
}

// --------------------------- spike + transpose: (C, N) flat -> Xt (N, C) ----

__global__ void transpose_spike(const float* __restrict__ x, const float* __restrict__ pos,
                                _Float16* __restrict__ out, int ncols, int dq) {
  __shared__ float t[16][17];
  int tx = threadIdx.x, ty = threadIdx.y;
  int n0 = blockIdx.x * 16, c0 = blockIdx.y * 16, tb = blockIdx.z;
  size_t ib = (size_t)tb * 256 * ncols;
  size_t ii = ib + (size_t)(c0 + ty) * ncols + n0 + tx;
  float v = x[ii] + pos[ii];
  float q = rintf(fminf(fmaxf(v, 0.f), 4.f));
  t[ty][tx] = dq ? rintf(q * 0.25f) : q * 0.25f;   // query path is double-quantized (binary)
  __syncthreads();
  out[(size_t)tb * ncols * 256 + (size_t)(n0 + ty) * 256 + c0 + tx] = (_Float16)t[tx][ty];
}

// ------------- GEMM orientation 1: D(lane=n, rows=o) -> store Out (N, C) ----
// Y = spike4(We @ X + be), X given as Xt rows (N x 256). Wave tile: 16o x 64n.

__global__ void gemm_o1(const _Float16* __restrict__ We, const float* __restrict__ be,
                        const _Float16* __restrict__ Xt, _Float16* __restrict__ Out, int N) {
  int w = threadIdx.x >> 5, lane = threadIdx.x & 31;
  int r = lane & 15, hi = lane >> 4;
  int nt64 = N >> 6;
  int wg = blockIdx.x * 8 + w;
  int per_tb = 16 * nt64;
  int tb = wg / per_tb, rem = wg % per_tb;
  int ot = rem / nt64, nt = rem % nt64;
  int o0 = ot * 16, n0 = nt * 64;

  const _Float16* wb = We + (size_t)o0 * 256;
  const _Float16* xb = Xt + (size_t)tb * N * 256 + (size_t)n0 * 256;

  v8f c[4];
#pragma unroll
  for (int rr = 0; rr < 8; ++rr) {
    float bv = be[o0 + rr + 8 * hi];
    c[0][rr] = bv; c[1][rr] = bv; c[2][rr] = bv; c[3][rr] = bv;
  }
#pragma unroll
  for (int ks = 0; ks < 8; ++ks) {
    int k0 = ks * 32;
    v16h a = load_frag(wb + k0, 256);
#pragma unroll
    for (int s2 = 0; s2 < 4; ++s2) {
      v16h b2 = load_frag(xb + (size_t)s2 * 16 * 256 + k0, 256);
      c[s2] = __builtin_amdgcn_wmma_f32_16x16x32_f16(false, a, false, b2,
                                                     (short)0, c[s2], false, false);
    }
  }
#pragma unroll
  for (int s2 = 0; s2 < 4; ++s2) {
    int n = n0 + s2 * 16 + r;
    v8h pk;
#pragma unroll
    for (int rr = 0; rr < 8; ++rr) pk[rr] = (_Float16)spike4(c[s2][rr]);
    *(v8h*)(Out + (size_t)tb * N * 256 + (size_t)n * 256 + o0 + 8 * hi) = pk;
  }
}

// ------------- GEMM orientation 2: D(lane=o, rows=n) -> store Out (C, N) ----

__global__ void gemm_o2(const _Float16* __restrict__ We, const float* __restrict__ be,
                        const _Float16* __restrict__ Xt, _Float16* __restrict__ Out, int N) {
  int w = threadIdx.x >> 5, lane = threadIdx.x & 31;
  int r = lane & 15, hi = lane >> 4;
  int nt64 = N >> 6;
  int wg = blockIdx.x * 8 + w;
  int per_tb = nt64 * 16;
  int tb = wg / per_tb, rem = wg % per_tb;
  int nt = rem / 16, ot = rem % 16;
  int o0 = ot * 16, n0 = nt * 64;

  const _Float16* xb = Xt + (size_t)tb * N * 256 + (size_t)n0 * 256;

  v8f c[4];
  float bv = be[o0 + r];
#pragma unroll
  for (int rr = 0; rr < 8; ++rr) { c[0][rr] = bv; c[1][rr] = bv; c[2][rr] = bv; c[3][rr] = bv; }

#pragma unroll
  for (int ks = 0; ks < 8; ++ks) {
    int k0 = ks * 32;
    v16h b2 = load_frag(We + (size_t)o0 * 256 + k0, 256);   // B[k, n=o] = W[o, k]
#pragma unroll
    for (int s2 = 0; s2 < 4; ++s2) {
      v16h a = load_frag(xb + (size_t)s2 * 16 * 256 + k0, 256);
      c[s2] = __builtin_amdgcn_wmma_f32_16x16x32_f16(false, a, false, b2,
                                                     (short)0, c[s2], false, false);
    }
  }
  int oo = o0 + r;
#pragma unroll
  for (int s2 = 0; s2 < 4; ++s2) {
    v8h pk;
#pragma unroll
    for (int rr = 0; rr < 8; ++rr) pk[rr] = (_Float16)spike4(c[s2][rr]);
    *(v8h*)(Out + (size_t)tb * 256 * N + (size_t)oo * N + n0 + s2 * 16 + 8 * hi) = pk;
  }
}

// ------------------------------- fused softmax-free attention per (tb, h) ---
// S = 0.1 * Qh^T Kh (k = DH = 32, one WMMA), O += S * Vh^T, flash-style.

__global__ void attention(const _Float16* __restrict__ Qt, const _Float16* __restrict__ Kt,
                          const _Float16* __restrict__ V, _Float16* __restrict__ Xat) {
  __shared__ _Float16 sS[8][16 * 40];     // per-wave 16x32 tile, padded rows
  int w = threadIdx.x >> 5, lane = threadIdx.x & 31;
  int r = lane & 15, hi = lane >> 4;
  int blk = blockIdx.x;                   // 128 = tb(8) * h(8) * qhalf(2)
  int qhalf = blk & 1, h = (blk >> 1) & 7, tb = blk >> 4;
  int q0 = qhalf * 128 + w * 16;
  _Float16* my = &sS[w][0];

  const _Float16* qbase = Qt + (size_t)tb * kNQ * 256 + (size_t)q0 * 256 + h * 32;
  const _Float16* kbase = Kt + (size_t)tb * kNK * 256 + h * 32;
  const _Float16* vbase = V + (size_t)tb * 256 * kNK + (size_t)(h * 32) * kNK;

  v16h aq = load_frag(qbase, 256);        // A[nq, d], loaded once
  v8f acc0 = {}, acc1 = {};
  v8f zero = {};

  for (int nk0 = 0; nk0 < kNK; nk0 += 32) {
    v16h bk0 = load_frag(kbase + (size_t)nk0 * 256, 256);
    v16h bk1 = load_frag(kbase + (size_t)(nk0 + 16) * 256, 256);
    v8f s0 = __builtin_amdgcn_wmma_f32_16x16x32_f16(false, aq, false, bk0, (short)0, zero, false, false);
    v8f s1 = __builtin_amdgcn_wmma_f32_16x16x32_f16(false, aq, false, bk1, (short)0, zero, false, false);
    // D layout (lane=nk col, rows=nq) -> LDS [nq][nk]; same-wave DS is in-order.
#pragma unroll
    for (int rr = 0; rr < 8; ++rr) {
      my[(rr + 8 * hi) * 40 + r]      = (_Float16)(0.1f * s0[rr]);
      my[(rr + 8 * hi) * 40 + 16 + r] = (_Float16)(0.1f * s1[rr]);
    }
    // Re-read as A fragment (lane=nq row, halves=nk)
    v8h a2lo = *(v8h*)(my + r * 40 + hi * 8);
    v8h a2hi = *(v8h*)(my + r * 40 + 16 + hi * 8);
    v16h a2 = frag_combine(a2lo, a2hi);
    v16h bv0 = load_frag(vbase + nk0, kNK);                     // B[nk, d 0..15]
    v16h bv1 = load_frag(vbase + (size_t)16 * kNK + nk0, kNK);  // B[nk, d 16..31]
    acc0 = __builtin_amdgcn_wmma_f32_16x16x32_f16(false, a2, false, bv0, (short)0, acc0, false, false);
    acc1 = __builtin_amdgcn_wmma_f32_16x16x32_f16(false, a2, false, bv1, (short)0, acc1, false, false);
  }
  // spike (attn_lif), transpose via LDS, store Xat (NQ, C) coalesced.
#pragma unroll
  for (int rr = 0; rr < 8; ++rr) {
    my[(rr + 8 * hi) * 40 + r]      = (_Float16)spike4(acc0[rr]);
    my[(rr + 8 * hi) * 40 + 16 + r] = (_Float16)spike4(acc1[rr]);
  }
  v8h r0 = *(v8h*)(my + r * 40 + hi * 16);
  v8h r1 = *(v8h*)(my + r * 40 + hi * 16 + 8);
  _Float16* ob = Xat + (size_t)tb * kNQ * 256 + (size_t)(q0 + r) * 256 + h * 32 + hi * 16;
  *(v8h*)ob = r0;
  *(v8h*)(ob + 8) = r1;
}

// --------- final conv_bn (P) + residual (query + query_pos), f32 output ----

__global__ void gemm_p(const _Float16* __restrict__ We, const float* __restrict__ be,
                       const _Float16* __restrict__ Xt, const float* __restrict__ query,
                       const float* __restrict__ qpos, float* __restrict__ out) {
  int w = threadIdx.x >> 5, lane = threadIdx.x & 31;
  int r = lane & 15, hi = lane >> 4;
  int wg = blockIdx.x * 8 + w;            // 512 waves: tb(8) * nt(4) * ot(16)
  int tb = wg / 64, rem = wg % 64;
  int nt = rem / 16, ot = rem % 16;
  int o0 = ot * 16, n0 = nt * 64;

  const _Float16* xb = Xt + (size_t)tb * 256 * 256 + (size_t)n0 * 256;
  v8f c[4];
  float bv = be[o0 + r];
#pragma unroll
  for (int rr = 0; rr < 8; ++rr) { c[0][rr] = bv; c[1][rr] = bv; c[2][rr] = bv; c[3][rr] = bv; }

#pragma unroll
  for (int ks = 0; ks < 8; ++ks) {
    int k0 = ks * 32;
    v16h b2 = load_frag(We + (size_t)o0 * 256 + k0, 256);
#pragma unroll
    for (int s2 = 0; s2 < 4; ++s2) {
      v16h a = load_frag(xb + (size_t)s2 * 16 * 256 + k0, 256);
      c[s2] = __builtin_amdgcn_wmma_f32_16x16x32_f16(false, a, false, b2,
                                                     (short)0, c[s2], false, false);
    }
  }
  int oo = o0 + r;
  size_t base = (size_t)tb * 65536 + (size_t)oo * 256;
#pragma unroll
  for (int s2 = 0; s2 < 4; ++s2) {
    int nb = n0 + s2 * 16 + 8 * hi;
#pragma unroll
    for (int ch = 0; ch < 8; ch += 4) {
      float4 qa = *(const float4*)(query + base + nb + ch);
      float4 pa = *(const float4*)(qpos + base + nb + ch);
      float4 ov;
      ov.x = c[s2][ch + 0] + qa.x + pa.x;
      ov.y = c[s2][ch + 1] + qa.y + pa.y;
      ov.z = c[s2][ch + 2] + qa.z + pa.z;
      ov.w = c[s2][ch + 3] + qa.w + pa.w;
      *(float4*)(out + base + nb + ch) = ov;
    }
  }
}

// -------------------------------------------------------------- launcher ---

extern "C" void kernel_launch(void* const* d_in, const int* in_sizes, int n_in,
                              void* d_out, int out_size, void* d_ws, size_t ws_size,
                              hipStream_t stream) {
  (void)in_sizes; (void)n_in; (void)out_size; (void)ws_size;
  const float* query = (const float*)d_in[0];
  const float* key   = (const float*)d_in[1];
  // d_in[2] (value) is ignored by the reference (value path reuses spiked key)
  const float* qpos  = (const float*)d_in[3];
  const float* kpos  = (const float*)d_in[4];
  const float* Wm[4] = { (const float*)d_in[5],  (const float*)d_in[9],
                         (const float*)d_in[13], (const float*)d_in[17] };
  const float* bm[4] = { (const float*)d_in[6],  (const float*)d_in[10],
                         (const float*)d_in[14], (const float*)d_in[18] };
  const float* sm[4] = { (const float*)d_in[7],  (const float*)d_in[11],
                         (const float*)d_in[15], (const float*)d_in[19] };
  const float* om[4] = { (const float*)d_in[8],  (const float*)d_in[12],
                         (const float*)d_in[16], (const float*)d_in[20] };
  float* out = (float*)d_out;

  char* p = (char*)d_ws;
  _Float16* We  = (_Float16*)p; p += (size_t)4 * 65536 * 2;        // folded weights (f16)
  float*    be  = (float*)p;    p += (size_t)4 * 256 * 4;          // folded bias
  _Float16* Xtq = (_Float16*)p; p += (size_t)kTB * 256 * 256 * 2;  // spiked q, (NQ,C)
  _Float16* Xtk = (_Float16*)p; p += (size_t)kTB * kNK * 256 * 2;  // spiked k, (NK,C)
  _Float16* Qt  = (_Float16*)p; p += (size_t)kTB * 256 * 256 * 2;  // Q, (NQ,C)
  _Float16* Kt  = (_Float16*)p; p += (size_t)kTB * kNK * 256 * 2;  // K, (NK,C)
  _Float16* Vm  = (_Float16*)p; p += (size_t)kTB * 256 * kNK * 2;  // V, (C,NK)
  _Float16* Xat = (_Float16*)p;                                    // attn out, (NQ,C)

  for (int m = 0; m < 4; ++m)
    prep_weights<<<256, 256, 0, stream>>>(Wm[m], bm[m], sm[m], om[m],
                                          We + (size_t)m * 65536, be + m * 256);

  transpose_spike<<<dim3(16, 16, kTB), dim3(16, 16), 0, stream>>>(query, qpos, Xtq, 256, 1);
  transpose_spike<<<dim3(256, 16, kTB), dim3(16, 16), 0, stream>>>(key, kpos, Xtk, 4096, 0);

  gemm_o1<<<64,   256, 0, stream>>>(We + 0 * 65536, be + 0,   Xtq, Qt, 256);   // Q
  gemm_o1<<<1024, 256, 0, stream>>>(We + 1 * 65536, be + 256, Xtk, Kt, 4096);  // K
  gemm_o2<<<1024, 256, 0, stream>>>(We + 2 * 65536, be + 512, Xtk, Vm, 4096);  // V

  attention<<<128, 256, 0, stream>>>(Qt, Kt, Vm, Xat);

  gemm_p<<<64, 256, 0, stream>>>(We + 3 * 65536, be + 768, Xat, query, qpos, out);
}
